// TransformerBlock_40149354283073
// MI455X (gfx1250) — matrix operand order repaired and checked by reading, not compiled, not run on hardware
//
#include <hip/hip_runtime.h>
#include <hip/hip_bf16.h>
#include <math.h>

// ---------------------------------------------------------------------------
// MI455X / gfx1250 transformer block.
// All matmuls use V_WMMA_F32_16X16X32_BF16 (bf16 in, fp32 accumulate).
// Softmax / RMSNorm / RoPE / residuals in fp32.
// Shapes fixed by the reference: B=1, S=2048, H=2048, NH=16, DH=128, FF=8192.
// ---------------------------------------------------------------------------

typedef __attribute__((ext_vector_type(16))) __bf16 bf16x16;
typedef __attribute__((ext_vector_type(8)))  __bf16 bf16x8;
typedef __attribute__((ext_vector_type(8)))  float  f32x8;

#define S_LEN  2048
#define H_DIM  2048
#define NHEAD  16
#define DHEAD  128
#define FF_DIM 8192

__device__ __forceinline__ bf16x16 bf_combine(bf16x8 lo, bf16x8 hi) {
  bf16x16 r;
#pragma unroll
  for (int i = 0; i < 8; i++) { r[i] = lo[i]; r[i + 8] = hi[i]; }
  return r;
}

// ---------------------------------------------------------------------------
// fp32 -> bf16 elementwise convert (weights / activations)
// ---------------------------------------------------------------------------
__global__ void cvt_f32_bf16(const float* __restrict__ in,
                             __bf16* __restrict__ out, size_t n) {
  size_t i = (size_t)blockIdx.x * 256 + threadIdx.x;
  if (i < n) out[i] = (__bf16)in[i];
}

// ---------------------------------------------------------------------------
// RMSNorm: one block per row, fp32 math, bf16 out
// ---------------------------------------------------------------------------
__global__ __launch_bounds__(256)
void rmsnorm_bf16(const float* __restrict__ x, const float* __restrict__ w,
                  __bf16* __restrict__ out, int H) {
  __shared__ float red[256];
  const int row = blockIdx.x, tid = threadIdx.x;
  const float* xr = x + (size_t)row * H;
  float s = 0.f;
  for (int j = tid; j < H; j += 256) { float v = xr[j]; s += v * v; }
  red[tid] = s;
  __syncthreads();
  for (int off = 128; off > 0; off >>= 1) {
    if (tid < off) red[tid] += red[tid + off];
    __syncthreads();
  }
  const float inv = rsqrtf(red[0] / (float)H + 1e-5f);
  for (int j = tid; j < H; j += 256)
    out[(size_t)row * H + j] = (__bf16)(w[j] * xr[j] * inv);
}

// ---------------------------------------------------------------------------
// RoPE: q' = q*cos(f) + rotate_half(q)*sin(f), fp32 in, bf16 out
// ---------------------------------------------------------------------------
__global__ void rope_bf16(const float* __restrict__ q,
                          const float* __restrict__ freqs,
                          __bf16* __restrict__ out, int S, int H) {
  int i = blockIdx.x * 256 + threadIdx.x;
  if (i >= S * H) return;
  const int s = i / H, c = i % H;
  const int d = c & (DHEAD - 1);
  const float f = freqs[(size_t)s * DHEAD + d];
  const float x = q[i];
  const float r = (d < DHEAD / 2) ? -q[i + DHEAD / 2] : q[i - DHEAD / 2];
  out[i] = (__bf16)(x * cosf(f) + r * sinf(f));
}

// ---------------------------------------------------------------------------
// Tiled bf16 GEMM: C[M,N] = A[M,K] @ B[K,N], both row-major bf16,
// fp32 accumulation via v_wmma_f32_16x16x32_bf16.
// Block = 128x128 tile, 8 waves (2x4), each wave 64x32 (4x2 WMMA frags).
// B tile is stored transposed in LDS so B fragments (same k order as A frags: lane n=N,
// 16 contiguous K values) are contiguous reads.
// MODE: 0 = store f32, 1 = store bf16, 2 = store f32 resid[idx]+acc,
//       3 = store bf16( silu(aux[idx]) * acc )   (SwiGLU fuse)
// ---------------------------------------------------------------------------
template <int MODE>
__global__ __launch_bounds__(256)
void gemm_bf16(const __bf16* __restrict__ A, const __bf16* __restrict__ B,
               void* __restrict__ Cv, const float* __restrict__ aux,
               const float* __restrict__ resid, int M, int N, int K) {
  __shared__ __bf16 As[128][40];   // [m][k], pad to 40 (16B-aligned rows)
  __shared__ __bf16 Bs[128][40];   // [n][k] (transposed tile)

  const int tid  = threadIdx.x;
  const int lane = tid & 31, wave = tid >> 5;
  const int wm = wave >> 2, wn = wave & 3;       // wave grid 2 (M) x 4 (N)
  const int m0 = blockIdx.y * 128, n0 = blockIdx.x * 128;
  const int lhalf = lane >> 4, ln = lane & 15;

  f32x8 acc[4][2];
#pragma unroll
  for (int mi = 0; mi < 4; mi++)
#pragma unroll
    for (int ni = 0; ni < 2; ni++)
#pragma unroll
      for (int r = 0; r < 8; r++) acc[mi][ni][r] = 0.f;

  const int arow = tid >> 1, aseg = tid & 1;   // A loader: 128 rows x 2 segs
  const int brow = tid >> 3, bseg = tid & 7;   // B loader: 32 rows x 8 segs

  for (int k0 = 0; k0 < K; k0 += 32) {
    {  // A tile 128x32
      const __bf16* g = A + (size_t)(m0 + arow) * K + k0 + aseg * 16;
      *(bf16x8*)&As[arow][aseg * 16]     = *(const bf16x8*)g;
      *(bf16x8*)&As[arow][aseg * 16 + 8] = *(const bf16x8*)(g + 8);
    }
    {  // B tile 32x128 -> Bs[n][k]
      const __bf16* g = B + (size_t)(k0 + brow) * N + n0 + bseg * 16;
      bf16x8 lo = *(const bf16x8*)g;
      bf16x8 hi = *(const bf16x8*)(g + 8);
#pragma unroll
      for (int j = 0; j < 8; j++) Bs[bseg * 16 + j][brow] = lo[j];
#pragma unroll
      for (int j = 0; j < 8; j++) Bs[bseg * 16 + 8 + j][brow] = hi[j];
    }
    __syncthreads();

    bf16x16 afrag[4], bfrag[2];
#pragma unroll
    for (int mi = 0; mi < 4; mi++) {
      // A frag: m = lane&15, K = 16 contiguous at half*16
      const __bf16* p = &As[wm * 64 + mi * 16 + ln][lhalf * 16];
      afrag[mi] = bf_combine(*(const bf16x8*)p, *(const bf16x8*)(p + 8));
    }
#pragma unroll
    for (int ni = 0; ni < 2; ni++) {
      // B frag: n = lane&15, K = 16 contiguous at half*16
      const __bf16* p = &Bs[wn * 32 + ni * 16 + ln][lhalf * 16];
      bfrag[ni] = bf_combine(*(const bf16x8*)p, *(const bf16x8*)(p + 8));
    }
#pragma unroll
    for (int mi = 0; mi < 4; mi++)
#pragma unroll
      for (int ni = 0; ni < 2; ni++)
        acc[mi][ni] = __builtin_amdgcn_wmma_f32_16x16x32_bf16(
            false, afrag[mi], false, bfrag[ni], (short)0, acc[mi][ni],
            false, false);
    __syncthreads();
  }

  // Epilogue. C frag layout: VGPR r -> row r (+8 for lanes 16..31), col = lane&15.
#pragma unroll
  for (int mi = 0; mi < 4; mi++)
#pragma unroll
    for (int ni = 0; ni < 2; ni++)
#pragma unroll
      for (int r = 0; r < 8; r++) {
        const int row = m0 + wm * 64 + mi * 16 + lhalf * 8 + r;
        const int col = n0 + wn * 32 + ni * 16 + ln;
        const size_t idx = (size_t)row * N + col;
        const float v = acc[mi][ni][r];
        if (MODE == 0) {
          ((float*)Cv)[idx] = v;
        } else if (MODE == 1) {
          ((__bf16*)Cv)[idx] = (__bf16)v;
        } else if (MODE == 2) {
          ((float*)Cv)[idx] = resid[idx] + v;
        } else {
          const float g = aux[idx];
          ((__bf16*)Cv)[idx] = (__bf16)((g / (1.0f + __expf(-g))) * v);
        }
      }
}

// ---------------------------------------------------------------------------
// Flash attention: block = (head, 128 query rows), 8 waves x 16 rows.
// scores = (Q Kt)/sqrt(DH) + mask + alibi ; online softmax ; O = P V.
// Causal tiles above the diagonal are skipped entirely.
// LDS: Ks[128][136] (K tile, == Bt for QKt), Vs[128][136] (V transposed),
//      Ps[8][16][136] (per-wave P staging for the A-operand relayout).
// ---------------------------------------------------------------------------
__global__ __launch_bounds__(256)
void flash_attn(const __bf16* __restrict__ qb, const __bf16* __restrict__ kb,
                const __bf16* __restrict__ vb, const float* __restrict__ mask,
                const float* __restrict__ alibi, __bf16* __restrict__ ob) {
  extern __shared__ __bf16 smem[];
  __bf16* Ks = smem;                  // 128*136
  __bf16* Vs = smem + 128 * 136;      // 128*136
  __bf16* Ps = smem + 2 * 128 * 136;  // 8*16*136

  const int S = S_LEN, H = H_DIM;
  const int qt = blockIdx.x, h = blockIdx.y;
  const int q0 = qt * 128;
  const int tid = threadIdx.x, lane = tid & 31, wave = tid >> 5;
  const int lhalf = lane >> 4, ln = lane & 15;

  // Q fragments for this wave's 16 rows (held in registers across kv loop)
  bf16x16 qf[4];
  {
    const __bf16* qrow = qb + (size_t)(q0 + wave * 16 + ln) * H + h * DHEAD;
#pragma unroll
    for (int kk = 0; kk < 4; kk++) {
      const __bf16* p = qrow + kk * 32 + lhalf * 16;
      qf[kk] = bf_combine(*(const bf16x8*)p, *(const bf16x8*)(p + 8));
    }
  }

  f32x8 o[8];
#pragma unroll
  for (int nt = 0; nt < 8; nt++)
#pragma unroll
    for (int r = 0; r < 8; r++) o[nt][r] = 0.f;
  float mrow[8], lrow[8];
#pragma unroll
  for (int r = 0; r < 8; r++) { mrow[r] = -3.0e38f; lrow[r] = 0.f; }

  const float scl = 0.08838834764831845f;  // 1/sqrt(128)

  for (int t = 0; t <= qt; t++) {          // causal: skip tiles past diagonal
    const int kv0 = t * 128;
    {  // stage K (as-is) and V (transposed) tiles
      const int r = tid >> 1, half = tid & 1;
      const __bf16* kg = kb + (size_t)(kv0 + r) * H + h * DHEAD + half * 64;
      __bf16* kd = Ks + r * 136 + half * 64;
#pragma unroll
      for (int j = 0; j < 8; j++)
        *(bf16x8*)(kd + j * 8) = *(const bf16x8*)(kg + j * 8);
      const __bf16* vg = vb + (size_t)(kv0 + r) * H + h * DHEAD + half * 64;
#pragma unroll
      for (int j = 0; j < 8; j++) {
        bf16x8 v8 = *(const bf16x8*)(vg + j * 8);
#pragma unroll
        for (int e = 0; e < 8; e++) Vs[(half * 64 + j * 8 + e) * 136 + r] = v8[e];
      }
    }
    __syncthreads();

    // ---- scores: S = Q @ Kt (Bt == K tile rows, contiguous in Ks) ----
    f32x8 sc[8];
#pragma unroll
    for (int nt = 0; nt < 8; nt++) {
#pragma unroll
      for (int r = 0; r < 8; r++) sc[nt][r] = 0.f;
#pragma unroll
      for (int kk = 0; kk < 4; kk++) {
        const __bf16* p = Ks + (nt * 16 + ln) * 136 + kk * 32 + lhalf * 16;
        bf16x16 bf = bf_combine(*(const bf16x8*)p, *(const bf16x8*)(p + 8));
        sc[nt] = __builtin_amdgcn_wmma_f32_16x16x32_bf16(
            false, qf[kk], false, bf, (short)0, sc[nt], false, false);
      }
    }

    // ---- scale + mask + alibi, online softmax ----
    const int rowbase = q0 + wave * 16 + lhalf * 8;
    float mloc[8];
#pragma unroll
    for (int r = 0; r < 8; r++) mloc[r] = -3.0e38f;
#pragma unroll
    for (int nt = 0; nt < 8; nt++) {
      const int col = kv0 + nt * 16 + ln;
#pragma unroll
      for (int r = 0; r < 8; r++) {
        const int row = rowbase + r;
        const float v = sc[nt][r] * scl + mask[(size_t)row * S + col] +
                        alibi[((size_t)h * S + row) * S + col];
        sc[nt][r] = v;
        mloc[r] = fmaxf(mloc[r], v);
      }
    }
#pragma unroll
    for (int off = 1; off < 16; off <<= 1)
#pragma unroll
      for (int r = 0; r < 8; r++)
        mloc[r] = fmaxf(mloc[r], __shfl_xor(mloc[r], off, 32));

    float sume[8];
#pragma unroll
    for (int r = 0; r < 8; r++) {
      const float mn = fmaxf(mrow[r], mloc[r]);
      const float rs = __expf(mrow[r] - mn);
      mrow[r] = mn;
      lrow[r] *= rs;
      sume[r] = 0.f;
#pragma unroll
      for (int nt = 0; nt < 8; nt++) o[nt][r] *= rs;
    }
#pragma unroll
    for (int nt = 0; nt < 8; nt++)
#pragma unroll
      for (int r = 0; r < 8; r++) {
        const float p = __expf(sc[nt][r] - mrow[r]);
        sume[r] += p;
        Ps[(wave * 16 + lhalf * 8 + r) * 136 + nt * 16 + ln] = (__bf16)p;
      }
#pragma unroll
    for (int off = 1; off < 16; off <<= 1)
#pragma unroll
      for (int r = 0; r < 8; r++) sume[r] += __shfl_xor(sume[r], off, 32);
#pragma unroll
    for (int r = 0; r < 8; r++) lrow[r] += sume[r];

    __syncthreads();  // P visible (cross-lane relayout through LDS)

    // ---- O += P @ V (Vs is [dh][kv] -> contiguous B fragments) ----
#pragma unroll
    for (int kk = 0; kk < 4; kk++) {
      const __bf16* pp = Ps + (wave * 16 + ln) * 136 + kk * 32 + lhalf * 16;
      bf16x16 pf = bf_combine(*(const bf16x8*)pp, *(const bf16x8*)(pp + 8));
#pragma unroll
      for (int nt = 0; nt < 8; nt++) {
        const __bf16* vp = Vs + (nt * 16 + ln) * 136 + kk * 32 + lhalf * 16;
        bf16x16 vf = bf_combine(*(const bf16x8*)vp, *(const bf16x8*)(vp + 8));
        o[nt] = __builtin_amdgcn_wmma_f32_16x16x32_bf16(
            false, pf, false, vf, (short)0, o[nt], false, false);
      }
    }
    __syncthreads();  // before next iteration overwrites Ks/Vs
  }

  // ---- finalize: O /= l, write bf16 [S,H] (head columns) ----
#pragma unroll
  for (int r = 0; r < 8; r++) {
    const float inv = 1.0f / lrow[r];
    const int row = q0 + wave * 16 + lhalf * 8 + r;
#pragma unroll
    for (int nt = 0; nt < 8; nt++)
      ob[(size_t)row * H + h * DHEAD + nt * 16 + ln] = (__bf16)(o[nt][r] * inv);
  }
}

// ---------------------------------------------------------------------------
// Host orchestration (graph-capture safe: only async launches on `stream`)
// ---------------------------------------------------------------------------
extern "C" void kernel_launch(void* const* d_in, const int* in_sizes, int n_in,
                              void* d_out, int out_size, void* d_ws,
                              size_t ws_size, hipStream_t stream) {
  const int S = S_LEN, H = H_DIM, FF = FF_DIM, NH = NHEAD;
  const float* x     = (const float*)d_in[0];
  const float* mask  = (const float*)d_in[1];
  const float* alibi = (const float*)d_in[2];
  const float* freqs = (const float*)d_in[3];
  const float* wq    = (const float*)d_in[4];
  const float* wk    = (const float*)d_in[5];
  const float* wv    = (const float*)d_in[6];
  const float* wo    = (const float*)d_in[7];
  const float* w1    = (const float*)d_in[8];
  const float* w2    = (const float*)d_in[9];   // dict order: w2 before w3
  const float* w3    = (const float*)d_in[10];
  const float* anw   = (const float*)d_in[11];
  const float* fnw   = (const float*)d_in[12];
  float* out = (float*)d_out;

  // ---- workspace carve (320 MiB) ----
  char* ws = (char*)d_ws;
  size_t off = 0;
  auto carve = [&](size_t bytes) -> char* {
    char* p = ws + off;
    off += (bytes + 255) & ~(size_t)255;
    return p;
  };
  const size_t HH = (size_t)H * H, HF = (size_t)H * FF, SH = (size_t)S * H,
               SF = (size_t)S * FF;
  __bf16* nx_bf  = (__bf16*)carve(SH * 2);
  __bf16* wq_bf  = (__bf16*)carve(HH * 2);
  __bf16* wk_bf  = (__bf16*)carve(HH * 2);
  __bf16* wv_bf  = (__bf16*)carve(HH * 2);
  __bf16* wo_bf  = (__bf16*)carve(HH * 2);
  __bf16* w1_bf  = (__bf16*)carve(HF * 2);
  __bf16* w3_bf  = (__bf16*)carve(HF * 2);
  __bf16* w2_bf  = (__bf16*)carve(HF * 2);
  float*  q_f32  = (float*)carve(SH * 4);
  float*  k_f32  = (float*)carve(SH * 4);
  __bf16* q_bf   = (__bf16*)carve(SH * 2);
  __bf16* k_bf   = (__bf16*)carve(SH * 2);
  __bf16* v_bf   = (__bf16*)carve(SH * 2);
  __bf16* at_bf  = (__bf16*)carve(SH * 2);
  float*  x1     = (float*)carve(SH * 4);
  __bf16* nx2_bf = (__bf16*)carve(SH * 2);
  float*  g1_f32 = (float*)carve(SF * 4);
  __bf16* h_bf   = (__bf16*)carve(SF * 2);
  (void)ws_size; (void)in_sizes; (void)n_in; (void)out_size;

  auto cvt = [&](const float* s, __bf16* d, size_t n) {
    cvt_f32_bf16<<<(unsigned)((n + 255) / 256), 256, 0, stream>>>(s, d, n);
  };
  cvt(wq, wq_bf, HH); cvt(wk, wk_bf, HH); cvt(wv, wv_bf, HH); cvt(wo, wo_bf, HH);
  cvt(w1, w1_bf, HF); cvt(w3, w3_bf, HF); cvt(w2, w2_bf, HF);

  // 1) attention-input RMSNorm
  rmsnorm_bf16<<<S, 256, 0, stream>>>(x, anw, nx_bf, H);

  // 2) Q/K/V projections (bf16 WMMA GEMMs)
  dim3 gH(H / 128, S / 128), gF(FF / 128, S / 128);
  gemm_bf16<0><<<gH, 256, 0, stream>>>(nx_bf, wq_bf, q_f32, nullptr, nullptr, S, H, H);
  gemm_bf16<0><<<gH, 256, 0, stream>>>(nx_bf, wk_bf, k_f32, nullptr, nullptr, S, H, H);
  gemm_bf16<1><<<gH, 256, 0, stream>>>(nx_bf, wv_bf, v_bf,  nullptr, nullptr, S, H, H);

  // 3) RoPE on Q and K (fp32 math -> bf16)
  rope_bf16<<<(S * H + 255) / 256, 256, 0, stream>>>(q_f32, freqs, q_bf, S, H);
  rope_bf16<<<(S * H + 255) / 256, 256, 0, stream>>>(k_f32, freqs, k_bf, S, H);

  // 4) fused causal attention with alibi (flash-style, no S x S tensor)
  const size_t attn_lds = (size_t)(2 * 128 * 136 + 8 * 16 * 136) * sizeof(__bf16);
  flash_attn<<<dim3(S / 128, NH), 256, attn_lds, stream>>>(q_bf, k_bf, v_bf,
                                                           mask, alibi, at_bf);

  // 5) output projection + residual: x1 = x + attn @ wo
  gemm_bf16<2><<<gH, 256, 0, stream>>>(at_bf, wo_bf, x1, nullptr, x, S, H, H);

  // 6) FFN: SwiGLU fused into second GEMM's epilogue
  rmsnorm_bf16<<<S, 256, 0, stream>>>(x1, fnw, nx2_bf, H);
  gemm_bf16<0><<<gF, 256, 0, stream>>>(nx2_bf, w1_bf, g1_f32, nullptr, nullptr, S, FF, H);
  gemm_bf16<3><<<gF, 256, 0, stream>>>(nx2_bf, w3_bf, h_bf, g1_f32, nullptr, S, FF, H);

  // 7) down-projection + residual -> d_out
  gemm_bf16<2><<<gH, 256, 0, stream>>>(h_bf, w2_bf, out, nullptr, x1, S, H, FF);
}
